// HybridMambaTransformerEncoder_58007828299855
// MI455X (gfx1250) — matrix operand order repaired and checked by reading, not compiled, not run on hardware
//
#include <hip/hip_runtime.h>
#include <hip/hip_bf16.h>
#include <math.h>

// ---------------- problem constants ----------------
#define Bsz     8
#define Lseq    1024
#define Dmodel  512
#define DI      1024          // EXPAND * D_MODEL
#define DTRANK  32
#define DSTATE  16
#define DCONV   4
#define NHEAD   8
#define DH      64
#define DFF     2048
#define Mrows   (Bsz * Lseq)  // 8192

typedef __attribute__((ext_vector_type(16))) _Float16 v16h;
typedef __attribute__((ext_vector_type(8)))  float    v8f;

// ---------------- WMMA GEMM:  C = act(scale * A @ op(B) + bias) ----------------
// A: (M,K) row-major fp32, lda
// B: transB=1 -> (N,K) row-major ; transB=0 -> (K,N) row-major
// batched over blockIdx.z with per-z element offsets aB/bB/cB.
// Requires M%128==0, N%64==0, K%32==0.
// Block tile: 128x64, 4 waves; wave w computes rows [w*32, w*32+32) x all 64 cols
// = 2x4 WMMA fragments = 8 v_wmma per K-step.
struct GemmArgs {
  const float* A; const float* B; const float* bias; float* C;
  int M, N, K;
  int lda, ldb, ldc;
  long aB, bB, cB;
  float scale;
  int transB;
  int act; // 0 none, 1 relu, 2 silu, 3 softplus
};

__global__ __launch_bounds__(128) void gemm_wmma_f16(GemmArgs g) {
  const float* A  = g.A + (long)blockIdx.z * g.aB;
  const float* Bp = g.B + (long)blockIdx.z * g.bB;
  float*       C  = g.C + (long)blockIdx.z * g.cB;

  // Both tiles stored with K contiguous (row-major over [m|n][k]) so that both
  // A-fragment (k = e + 16*hi) and B-fragment (k = e + 16*hi) loads
  // are contiguous 16B/32B runs -> ds_load_b128.
  __shared__ _Float16 As[128][40];   // 128 x 32 (+pad)
  __shared__ _Float16 Bs[64][40];    //  64 x 32 (+pad)

  const int tid  = threadIdx.x;
  const int wave = tid >> 5;
  const int lane = tid & 31;
  const int ml = lane & 15, hi = lane >> 4;

  const int m0 = blockIdx.y * 128;
  const int n0 = blockIdx.x * 64;

  v8f acc[2][4] = {};

  for (int k0 = 0; k0 < g.K; k0 += 32) {
    // ---- stage A tile (128x32): each thread converts one full row ----
    {
      const float* src = A + (long)(m0 + tid) * g.lda + k0;
#pragma unroll
      for (int c = 0; c < 32; ++c) As[tid][c] = (_Float16)src[c];
    }
    // ---- stage B tile (64x32, K contiguous) ----
    if (g.transB) {
      const int r  = tid >> 1;            // n within tile
      const int c0 = (tid & 1) * 16;      // k half
      const float* src = Bp + (long)(n0 + r) * g.ldb + (k0 + c0);
#pragma unroll
      for (int i = 0; i < 16; ++i) Bs[r][c0 + i] = (_Float16)src[i];
    } else {
      // B is (K,N): coalesced global reads, 16-bit LDS transpose scatter
#pragma unroll
      for (int i = 0; i < 16; ++i) {
        int idx = tid * 16 + i;
        int kk = idx >> 6, nn = idx & 63;
        Bs[nn][kk] = (_Float16)Bp[(long)(k0 + kk) * g.ldb + (n0 + nn)];
      }
    }
    // prefetch next K tile (global_prefetch_b8)
    if (k0 + 32 < g.K) {
      __builtin_prefetch(&A[(long)(m0 + tid) * g.lda + (k0 + 32)], 0, 1);
      __builtin_prefetch(&Bp[(long)(tid >> 1) * g.ldb + (k0 + 32)], 0, 1);
    }
    __syncthreads();

    // ---- load fragments (contiguous k = e + 16*hi, same order for A and B) ----
    v16h af[2], bf[4];
#pragma unroll
    for (int s = 0; s < 2; ++s) {
      const int row = wave * 32 + s * 16 + ml;
#pragma unroll
      for (int e = 0; e < 16; ++e) af[s][e] = As[row][e + 16 * hi];      // k = e + 16*hi
    }
#pragma unroll
    for (int j = 0; j < 4; ++j) {
      const int col = j * 16 + ml;
#pragma unroll
      for (int e = 0; e < 16; ++e) bf[j][e] = Bs[col][e + 16 * hi];      // k = e + 16*hi
    }

#pragma unroll
    for (int s = 0; s < 2; ++s)
#pragma unroll
      for (int j = 0; j < 4; ++j)
        acc[s][j] = __builtin_amdgcn_wmma_f32_16x16x32_f16(
            false, af[s], false, bf[j], (short)0, acc[s][j], false, false);
    __syncthreads();
  }

  // ---- epilogue ----
#pragma unroll
  for (int s = 0; s < 2; ++s) {
#pragma unroll
    for (int j = 0; j < 4; ++j) {
      const int nn = n0 + j * 16 + ml;
      const float bv = g.bias ? g.bias[nn] : 0.0f;
#pragma unroll
      for (int r = 0; r < 8; ++r) {
        const int mm = m0 + wave * 32 + s * 16 + r + 8 * hi;
        float v = acc[s][j][r] * g.scale + bv;
        if (g.act == 1)      v = fmaxf(v, 0.0f);
        else if (g.act == 2) v = v / (1.0f + __expf(-v));
        else if (g.act == 3) v = (v > 20.0f) ? v : log1pf(__expf(v));
        C[(long)mm * g.ldc + nn] = v;
      }
    }
  }
}

// ---------------- input projection: h[b,l,d] = sum_c x[b,c,l]*pw[d,c] + pb[d] ----------------
__global__ __launch_bounds__(256) void input_proj(const float* x, const float* pw,
                                                  const float* pb, float* h) {
  long idx = (long)blockIdx.x * 256 + threadIdx.x;     // over Mrows*Dmodel
  int d = idx & 511;
  long bl = idx >> 9;
  int b = (int)(bl >> 10);
  int l = (int)(bl & 1023);
  const float* xb = x + (long)b * 3 * Lseq;
  float acc = pb[d];
#pragma unroll
  for (int c = 0; c < 3; ++c) acc += xb[c * Lseq + l] * pw[d * 3 + c];
  h[idx] = acc;
}

// ---------------- depthwise causal conv (D_CONV=4) + SiLU ----------------
__global__ __launch_bounds__(256) void conv_silu(const float* xz, const float* cw,
                                                 const float* cb, float* xc) {
  long idx = (long)blockIdx.x * 256 + threadIdx.x;     // over Mrows*DI
  int di = (int)(idx & 1023);
  long bl = idx >> 10;
  int l = (int)(bl & 1023);
  float acc = cb[di];
#pragma unroll
  for (int j = 0; j < DCONV; ++j) {
    int ls = l - (DCONV - 1) + j;
    if (ls >= 0) acc += xz[(bl - l + ls) * (2 * DI) + di] * cw[di * DCONV + j];
  }
  xc[idx] = acc / (1.0f + __expf(-acc));
}

// ---------------- selective scan + D skip + SiLU(z) gate ----------------
__global__ __launch_bounds__(256) void mamba_scan_gate(const float* dt, const float* xdb,
                                                       const float* xc, const float* xz,
                                                       const float* Alog, const float* Dvec,
                                                       float* y) {
  int idx = blockIdx.x * 256 + threadIdx.x;            // 0..Bsz*DI-1
  int b = idx >> 10;
  int di = idx & 1023;

  float Arow[DSTATE];
#pragma unroll
  for (int s = 0; s < DSTATE; ++s) Arow[s] = -__expf(Alog[di * DSTATE + s]);
  const float Dv = Dvec[di];

  float st[DSTATE];
#pragma unroll
  for (int s = 0; s < DSTATE; ++s) st[s] = 0.0f;

  for (int l = 0; l < Lseq; ++l) {
    long base = (long)b * Lseq + l;
    float dtv = dt[base * DI + di];
    float xv  = xc[base * DI + di];
    const float* xb = xdb + base * (DTRANK + 2 * DSTATE);
    float dx = dtv * xv;
    float yv = 0.0f;
#pragma unroll
    for (int s = 0; s < DSTATE; ++s) {
      float dA = __expf(dtv * Arow[s]);
      st[s] = st[s] * dA + dx * xb[DTRANK + s];
      yv += st[s] * xb[DTRANK + DSTATE + s];
    }
    float zv = xz[base * (2 * DI) + DI + di];
    yv = (yv + Dv * xv) * (zv / (1.0f + __expf(-zv)));
    y[base * DI + di] = yv;
  }
}

// ---------------- out = LN(x (+ res)) * w + b,  row width 512 ----------------
__global__ __launch_bounds__(256) void add_ln(const float* x, const float* res,
                                              const float* w, const float* bi, float* out) {
  long row = blockIdx.x;
  int t = threadIdx.x;
  float v0 = x[row * 512 + t];
  float v1 = x[row * 512 + t + 256];
  if (res) { v0 += res[row * 512 + t]; v1 += res[row * 512 + t + 256]; }

  __shared__ float rs[256], rq[256];
  rs[t] = v0 + v1;
  rq[t] = v0 * v0 + v1 * v1;
  __syncthreads();
  for (int s = 128; s > 0; s >>= 1) {
    if (t < s) { rs[t] += rs[t + s]; rq[t] += rq[t + s]; }
    __syncthreads();
  }
  float mean = rs[0] * (1.0f / 512.0f);
  float var  = rq[0] * (1.0f / 512.0f) - mean * mean;
  float inv  = rsqrtf(var + 1e-5f);
  out[row * 512 + t]       = (v0 - mean) * inv * w[t] + bi[t];
  out[row * 512 + t + 256] = (v1 - mean) * inv * w[t + 256] + bi[t + 256];
}

// ---------------- sinusoidal positional encoding add ----------------
__global__ __launch_bounds__(256) void pos_add(float* h) {
  long idx = (long)blockIdx.x * 256 + threadIdx.x;     // Mrows*Dmodel
  int d = (int)(idx & 511);
  int l = (int)((idx >> 9) & 1023);
  float div = __expf(-(float)(d & ~1) * (9.210340371976184f / 512.0f)); // ln(1e4)/D
  float ang = (float)l * div;
  h[idx] += (d & 1) ? __cosf(ang) : __sinf(ang);
}

// ---------------- row softmax, width 1024 ----------------
__global__ __launch_bounds__(256) void softmax1024(float* S) {
  long row = blockIdx.x;
  float* p = S + row * 1024;
  int t = threadIdx.x;
  float v[4];
  float mx = -1e30f;
#pragma unroll
  for (int i = 0; i < 4; ++i) { v[i] = p[t + 256 * i]; mx = fmaxf(mx, v[i]); }
  __shared__ float red[256];
  red[t] = mx;
  __syncthreads();
  for (int s = 128; s > 0; s >>= 1) { if (t < s) red[t] = fmaxf(red[t], red[t + s]); __syncthreads(); }
  mx = red[0];
  __syncthreads();
  float sum = 0.0f;
#pragma unroll
  for (int i = 0; i < 4; ++i) { v[i] = __expf(v[i] - mx); sum += v[i]; }
  red[t] = sum;
  __syncthreads();
  for (int s = 128; s > 0; s >>= 1) { if (t < s) red[t] += red[t + s]; __syncthreads(); }
  float inv = 1.0f / red[0];
#pragma unroll
  for (int i = 0; i < 4; ++i) p[t + 256 * i] = v[i] * inv;
}

// ---------------- mean over L ----------------
__global__ __launch_bounds__(256) void mean_over_l(const float* h, float* out) {
  int idx = blockIdx.x * 256 + threadIdx.x;            // Bsz*Dmodel
  int d = idx & 511;
  int b = idx >> 9;
  float s = 0.0f;
  for (int l = 0; l < Lseq; ++l) s += h[((long)b * Lseq + l) * 512 + d];
  out[idx] = s * (1.0f / (float)Lseq);
}

// ---------------- host-side GEMM launcher ----------------
static void gemm(hipStream_t st, const float* A, const float* B, const float* bias, float* C,
                 int M, int N, int K, int lda, int ldb, int ldc,
                 int transB, int act, float scale,
                 int zcount = 1, long aB = 0, long bB = 0, long cB = 0) {
  GemmArgs g{A, B, bias, C, M, N, K, lda, ldb, ldc, aB, bB, cB, scale, transB, act};
  dim3 grid((unsigned)(N / 64), (unsigned)(M / 128), (unsigned)zcount);
  hipLaunchKernelGGL(gemm_wmma_f16, grid, dim3(128), 0, st, g);
}

extern "C" void kernel_launch(void* const* d_in, const int* in_sizes, int n_in,
                              void* d_out, int out_size, void* d_ws, size_t ws_size,
                              hipStream_t stream) {
  (void)in_sizes; (void)n_in; (void)out_size; (void)ws_size;

  const float* x        = (const float*)d_in[0];
  const float* proj_w   = (const float*)d_in[1];
  const float* proj_b   = (const float*)d_in[2];
  const float* m_in_w   = (const float*)d_in[3];
  const float* m_conv_w = (const float*)d_in[4];
  const float* m_conv_b = (const float*)d_in[5];
  const float* m_xp_w   = (const float*)d_in[6];
  const float* m_dt_w   = (const float*)d_in[7];
  const float* m_dt_b   = (const float*)d_in[8];
  const float* m_Alog   = (const float*)d_in[9];
  const float* m_D      = (const float*)d_in[10];
  const float* m_out_w  = (const float*)d_in[11];
  const float* m_ln_w   = (const float*)d_in[12];
  const float* m_ln_b   = (const float*)d_in[13];
  const float* t_qkv_w  = (const float*)d_in[14];
  const float* t_qkv_b  = (const float*)d_in[15];
  const float* t_ow     = (const float*)d_in[16];
  const float* t_ob     = (const float*)d_in[17];
  const float* t_l1w    = (const float*)d_in[18];
  const float* t_l1b    = (const float*)d_in[19];
  const float* t_l2w    = (const float*)d_in[20];
  const float* t_l2b    = (const float*)d_in[21];
  const float* t_ln1w   = (const float*)d_in[22];
  const float* t_ln1b   = (const float*)d_in[23];
  const float* t_ln2w   = (const float*)d_in[24];
  const float* t_ln2b   = (const float*)d_in[25];
  const float* no_w     = (const float*)d_in[26];
  const float* no_b     = (const float*)d_in[27];

  // ---- workspace layout (floats) ----
  float* p = (float*)d_ws;
  float* h       = p; p += (long)Mrows * Dmodel;              // current hidden
  float* tmp512  = p; p += (long)Mrows * Dmodel;              // proj results pre-LN
  float* buf2048 = p; p += (long)Mrows * 2048;                // xz / qkv / ff1
  float* xc      = p; p += (long)Mrows * DI;
  float* dtb     = p; p += (long)Mrows * DI;
  float* ybuf    = p; p += (long)Mrows * DI;                  // scan out & attn ctx
  float* xdb     = p; p += (long)Mrows * (DTRANK + 2 * DSTATE);
  float* scores  = p; p += (long)NHEAD * Lseq * Lseq;

  const int ELT = Mrows * Dmodel;  // 4,194,304

  // ---- input projection ----
  hipLaunchKernelGGL(input_proj, dim3(ELT / 256), dim3(256), 0, stream, x, proj_w, proj_b, h);

  // ---- Mamba layers ----
  for (int i = 0; i < 2; ++i) {
    // xz = h @ in_w^T : (8192, 2048)
    gemm(stream, h, m_in_w + (long)i * 2048 * 512, nullptr, buf2048,
         Mrows, 2048, 512, 512, 512, 2048, /*transB*/1, /*act*/0, 1.0f);
    // conv + silu -> xc
    hipLaunchKernelGGL(conv_silu, dim3((Mrows * DI) / 256), dim3(256), 0, stream,
                       buf2048, m_conv_w + (long)i * DI * DCONV, m_conv_b + (long)i * DI, xc);
    // xdb = xc @ xp_w^T : (8192, 64)
    gemm(stream, xc, m_xp_w + (long)i * 64 * DI, nullptr, xdb,
         Mrows, 64, DI, DI, DI, 64, 1, 0, 1.0f);
    // dt = softplus(xdb[:, :32] @ dt_w^T + dt_b) : (8192, 1024)
    gemm(stream, xdb, m_dt_w + (long)i * DI * DTRANK, m_dt_b + (long)i * DI, dtb,
         Mrows, DI, DTRANK, 64, DTRANK, DI, 1, /*softplus*/3, 1.0f);
    // selective scan + gate
    hipLaunchKernelGGL(mamba_scan_gate, dim3((Bsz * DI) / 256), dim3(256), 0, stream,
                       dtb, xdb, xc, buf2048,
                       m_Alog + (long)i * DI * DSTATE, m_D + (long)i * DI, ybuf);
    // out proj: (8192, 512)
    gemm(stream, ybuf, m_out_w + (long)i * 512 * DI, nullptr, tmp512,
         Mrows, 512, DI, DI, DI, 512, 1, 0, 1.0f);
    // h = LN(h + mamba_out)
    hipLaunchKernelGGL(add_ln, dim3(Mrows), dim3(256), 0, stream,
                       tmp512, h, m_ln_w + (long)i * 512, m_ln_b + (long)i * 512, h);
  }

  // ---- positional encoding ----
  hipLaunchKernelGGL(pos_add, dim3(ELT / 256), dim3(256), 0, stream, h);

  // ---- Transformer layers ----
  float* qkv = buf2048;   // (8192, 1536), alias (ff1 reuses buf2048 after qkv is dead)
  for (int i = 0; i < 2; ++i) {
    gemm(stream, h, t_qkv_w + (long)i * 1536 * 512, t_qkv_b + (long)i * 1536, qkv,
         Mrows, 1536, 512, 512, 512, 1536, 1, 0, 1.0f);

    for (int b = 0; b < Bsz; ++b) {
      const float* qb = qkv + (long)b * Lseq * 1536;
      // scores[h] = (q_h @ k_h^T) / sqrt(DH), heads batched on z
      gemm(stream, qb, qb + 512, nullptr, scores,
           Lseq, Lseq, DH, 1536, 1536, Lseq, /*transB*/1, 0, 0.125f,
           NHEAD, /*aB*/64, /*bB*/64, /*cB*/(long)Lseq * Lseq);
      hipLaunchKernelGGL(softmax1024, dim3(NHEAD * Lseq), dim3(256), 0, stream, scores);
      // ctx[h] = probs @ v_h : (1024, 64) written into (B,L,H,DH) layout
      gemm(stream, scores, qb + 1024, nullptr, ybuf + (long)b * Lseq * 512,
           Lseq, DH, Lseq, Lseq, 1536, 512, /*transB*/0, 0, 1.0f,
           NHEAD, /*aB*/(long)Lseq * Lseq, /*bB*/64, /*cB*/64);
    }
    // attn out proj
    gemm(stream, ybuf, t_ow + (long)i * 512 * 512, t_ob + (long)i * 512, tmp512,
         Mrows, 512, 512, 512, 512, 512, 1, 0, 1.0f);
    hipLaunchKernelGGL(add_ln, dim3(Mrows), dim3(256), 0, stream,
                       tmp512, h, t_ln1w + (long)i * 512, t_ln1b + (long)i * 512, h);
    // ff1 = relu(h @ l1w^T + b) : (8192, 2048)  (overwrites qkv region — qkv dead)
    gemm(stream, h, t_l1w + (long)i * DFF * 512, t_l1b + (long)i * DFF, buf2048,
         Mrows, DFF, 512, 512, 512, DFF, 1, /*relu*/1, 1.0f);
    // ff2 = ff1 @ l2w^T + b : (8192, 512)
    gemm(stream, buf2048, t_l2w + (long)i * 512 * DFF, t_l2b + (long)i * 512, tmp512,
         Mrows, 512, DFF, DFF, DFF, 512, 1, 0, 1.0f);
    hipLaunchKernelGGL(add_ln, dim3(Mrows), dim3(256), 0, stream,
                       tmp512, h, t_ln2w + (long)i * 512, t_ln2b + (long)i * 512, h);
  }

  // ---- final LN straight into d_out, then mean over L ----
  float* out_h = (float*)d_out;
  hipLaunchKernelGGL(add_ln, dim3(Mrows), dim3(256), 0, stream,
                     h, nullptr, no_w, no_b, out_h);
  hipLaunchKernelGGL(mean_over_l, dim3((Bsz * Dmodel) / 256), dim3(256), 0, stream,
                     out_h, out_h + (long)ELT);
}
